// WavenetResBlock_15874199126641
// MI455X (gfx1250) — matrix operand order repaired and checked by reading, not compiled, not run on hardware
//
#include <hip/hip_runtime.h>
#include <stdint.h>

// ---------------------------------------------------------------------------
// WaveNet residual block, fused, bf16 WMMA (gfx1250 / MI455X)
//   B=32, C=128, T=16000, K=2, dilation=2
//   out  = skip_conv(tanh(f_conv(x)) * sigmoid(g_conv(x))) + x
// GEMM view per conv tap: Y[co,t] = W[co,ci] @ X[ci,t]  (M=128,K=128,N=time)
// A = weights (resident in VGPRs, bf16), B = activations from LDS (bf16),
// accumulate f32 via v_wmma_f32_16x16x32_bf16.
// ---------------------------------------------------------------------------

typedef __attribute__((ext_vector_type(16))) __bf16    v16bf;
typedef __attribute__((ext_vector_type(8)))  float     v8f;
typedef __attribute__((ext_vector_type(8)))  uint32_t  v8u;

#define Bc      32
#define Cc      128
#define Tc      16000
#define TCHUNK  64
#define NCHUNK  (Bc * (Tc / TCHUNK))   // 8000
#define XP      68                     // xs pitch (dwords): 8*68 % 64 == 32 -> no bank clash
#define OP      84                     // os pitch (dwords): 8*84 % 64 == 32
#define BCT     ((size_t)Bc * Cc * Tc)

__device__ __forceinline__ uint16_t f2bf(float f) {
  uint32_t u = __builtin_bit_cast(uint32_t, f);
  u += 0x7FFFu + ((u >> 16) & 1u);               // round-to-nearest-even
  return (uint16_t)(u >> 16);
}

// A-matrix fragment (16x32 bf16, M=co, K=ci) from bf16 weights laid out [co][ci].
// Lanes 0-15 (m=lane): v0..3 = K k0..k0+7, v4..7 = K k0+8..k0+15
// Lanes 16-31        : v0..3 = K k0+16.., v4..7 = K k0+24..   (contiguous k = 16h + s, as B)
__device__ __forceinline__ v16bf load_afrag(const uint16_t* __restrict__ w, int elem_base) {
  uint4 q0 = *(const uint4*)(w + elem_base);
  uint4 q1 = *(const uint4*)(w + elem_base + 8);
  v8u a;
  a[0] = q0.x; a[1] = q0.y; a[2] = q0.z; a[3] = q0.w;
  a[4] = q1.x; a[5] = q1.y; a[6] = q1.z; a[7] = q1.w;
  return __builtin_bit_cast(v16bf, a);
}

// ---------------------------------------------------------------------------
// Prep: fp32 weights (co,ci,k) -> bf16 [conv][tap][co][ci] in workspace
// ---------------------------------------------------------------------------
__global__ void wavenet_prep(const float* __restrict__ wf,
                             const float* __restrict__ wg,
                             const float* __restrict__ ws,
                             uint16_t* __restrict__ wbf) {
  int i = blockIdx.x * 256 + threadIdx.x;          // 3*2*128*128 = 98304
  if (i >= 3 * 2 * Cc * Cc) return;
  int cv = i / (2 * Cc * Cc);
  int r  = i - cv * (2 * Cc * Cc);
  int tp = r / (Cc * Cc);
  int r2 = r - tp * (Cc * Cc);
  int co = r2 >> 7, ci = r2 & 127;
  const float* w = (cv == 0) ? wf : (cv == 1) ? wg : ws;
  wbf[i] = f2bf(w[(co * Cc + ci) * 2 + tp]);
}

// ---------------------------------------------------------------------------
// Main fused kernel. 256 threads = 8 waves; wave w owns output rows 16w..16w+15.
// ---------------------------------------------------------------------------
__global__ __launch_bounds__(256) void wavenet_main(
    const float* __restrict__ x, const uint16_t* __restrict__ wbf,
    float* __restrict__ out, float* __restrict__ skip) {
  __shared__ uint32_t xs[64 * XP];   // bf16 pairs (ci even|odd), origin t0-3, 67 cols
  __shared__ uint32_t os[64 * OP];   // bf16 pairs (co even|odd), origin t0-16, 80 cols

  const int tid  = threadIdx.x;
  const int wave = tid >> 5;
  const int lane = tid & 31;
  const int m    = lane & 15;
  const int hi   = lane >> 4;

  // Resident weight fragments: 3 convs x 2 taps x 4 k-tiles (24 x 8 VGPRs)
  v16bf Af[2][4], Ag[2][4], As[2][4];
#pragma unroll
  for (int tp = 0; tp < 2; ++tp)
#pragma unroll
    for (int kt = 0; kt < 4; ++kt) {
      int co   = wave * 16 + m;
      int base = co * Cc + kt * 32 + 16 * hi;
      Af[tp][kt] = load_afrag(wbf, (0 * 2 + tp) * Cc * Cc + base);
      Ag[tp][kt] = load_afrag(wbf, (1 * 2 + tp) * Cc * Cc + base);
      As[tp][kt] = load_afrag(wbf, (2 * 2 + tp) * Cc * Cc + base);
    }

  for (int chunk = blockIdx.x; chunk < NCHUNK; chunk += gridDim.x) {
    const int b  = chunk / (Tc / TCHUNK);
    const int t0 = (chunk - b * (Tc / TCHUNK)) * TCHUNK;

    // ---- stage x[b, :, t0-3 .. t0+63] -> LDS bf16 channel pairs --------
    for (int i = tid; i < 64 * 67; i += 256) {
      int p = i / 67, c = i - p * 67;
      int tg = t0 - 3 + c;
      float a0 = 0.f, a1 = 0.f;
      if (tg >= 0) {
        a0 = x[(size_t)(b * Cc + 2 * p)     * Tc + tg];
        a1 = x[(size_t)(b * Cc + 2 * p + 1) * Tc + tg];
      }
      xs[p * XP + c] = (uint32_t)f2bf(a0) | ((uint32_t)f2bf(a1) << 16);
    }
    __syncthreads();

    // ---- phase A: gated convs, 5 tiles over cols [t0-16, t0+64) --------
    for (int tt = 0; tt < 5; ++tt) {
      v8f yf = {}, yg = {};
#pragma unroll
      for (int kt = 0; kt < 4; ++kt) {
        int rowbase = (kt * 16 + 8 * hi) * XP;
        int c1 = 16 * tt + m - 13;                 // tap1: t        (xs origin t0-3)
        int c0 = c1 - 2;                           // tap0: t - dil
        c1 = c1 < 0 ? 0 : c1;                      // halo tile: clamp (cols unused)
        c0 = c0 < 0 ? 0 : c0;
        v8u b0v, b1v;
#pragma unroll
        for (int j = 0; j < 8; ++j) {
          b0v[j] = xs[rowbase + j * XP + c0];
          b1v[j] = xs[rowbase + j * XP + c1];
        }
        v16bf b0 = __builtin_bit_cast(v16bf, b0v);
        v16bf b1 = __builtin_bit_cast(v16bf, b1v);
        yf = __builtin_amdgcn_wmma_f32_16x16x32_bf16(false, Af[0][kt], false, b0, (short)0, yf, false, false);
        yf = __builtin_amdgcn_wmma_f32_16x16x32_bf16(false, Af[1][kt], false, b1, (short)0, yf, false, false);
        yg = __builtin_amdgcn_wmma_f32_16x16x32_bf16(false, Ag[0][kt], false, b0, (short)0, yg, false, false);
        yg = __builtin_amdgcn_wmma_f32_16x16x32_bf16(false, Ag[1][kt], false, b1, (short)0, yg, false, false);
      }
      // tanh(yf) * sigmoid(yg); D rows: lanes0-15 -> co=16w+j, lanes16-31 -> co=16w+8+j
      int col = 16 * tt + m;                       // os origin t0-16
#pragma unroll
      for (int jj = 0; jj < 4; ++jj) {
        float f0 = yf[2 * jj],     g0 = yg[2 * jj];
        float f1 = yf[2 * jj + 1], g1 = yg[2 * jj + 1];
        float e0 = __expf(-2.f * f0);
        float o0 = ((1.f - e0) / (1.f + e0)) * (1.f / (1.f + __expf(-g0)));
        float e1 = __expf(-2.f * f1);
        float o1 = ((1.f - e1) / (1.f + e1)) * (1.f / (1.f + __expf(-g1)));
        os[(wave * 8 + hi * 4 + jj) * OP + col] =
            (uint32_t)f2bf(o0) | ((uint32_t)f2bf(o1) << 16);
      }
    }
    __syncthreads();

    // ---- phase B: skip conv (+ residual), 4 tiles over [t0, t0+64) -----
    for (int tb = 0; tb < 4; ++tb) {
      v8f s = {};
#pragma unroll
      for (int kt = 0; kt < 4; ++kt) {
        int rowbase = (kt * 16 + 8 * hi) * OP;
        int c1 = 16 + 16 * tb + m;                 // tap1: t   (os origin t0-16)
        int c0 = c1 - 1;                           // tap0: t-1
        v8u b0v, b1v;
#pragma unroll
        for (int j = 0; j < 8; ++j) {
          b0v[j] = os[rowbase + j * OP + c0];
          b1v[j] = os[rowbase + j * OP + c1];
        }
        v16bf b0 = __builtin_bit_cast(v16bf, b0v);
        v16bf b1 = __builtin_bit_cast(v16bf, b1v);
        s = __builtin_amdgcn_wmma_f32_16x16x32_bf16(false, As[0][kt], false, b0, (short)0, s, false, false);
        s = __builtin_amdgcn_wmma_f32_16x16x32_bf16(false, As[1][kt], false, b1, (short)0, s, false, false);
      }
      int t = t0 + 16 * tb + m;
#pragma unroll
      for (int j = 0; j < 8; ++j) {
        int co = wave * 16 + 8 * hi + j;
        size_t idx = (size_t)(b * Cc + co) * Tc + t;
        float sv = s[j];
        out[idx]  = sv + x[idx];                   // residual in exact fp32
        skip[idx] = sv;
      }
    }
    __syncthreads();
  }
}

extern "C" void kernel_launch(void* const* d_in, const int* in_sizes, int n_in,
                              void* d_out, int out_size, void* d_ws, size_t ws_size,
                              hipStream_t stream) {
  const float* x  = (const float*)d_in[0];
  const float* wf = (const float*)d_in[1];
  const float* wg = (const float*)d_in[2];
  const float* ws = (const float*)d_in[3];
  // d_in[4] = dilation (==2, baked into the kernel)

  uint16_t* wbf = (uint16_t*)d_ws;                 // 196608 B of workspace
  float* out  = (float*)d_out;
  float* skip = out + BCT;

  wavenet_prep<<<(3 * 2 * Cc * Cc + 255) / 256, 256, 0, stream>>>(wf, wg, ws, wbf);
  wavenet_main<<<2000, 256, 0, stream>>>(x, wbf, out, skip);
}
